// CustomConv2d_90726889160758
// MI455X (gfx1250) — matrix operand order repaired and checked by reading, not compiled, not run on hardware
//
#include <hip/hip_runtime.h>

typedef _Float16 v16h __attribute__((ext_vector_type(16)));
typedef _Float16 v8h  __attribute__((ext_vector_type(8)));
typedef float    v8f  __attribute__((ext_vector_type(8)));
typedef int      g4i  __attribute__((vector_size(16)));   // <4 x i32> for async-LDS builtin

#define NB      32
#define CIN     128
#define COUT    256
#define HW      56
#define HW2     (HW * HW)            // 3136
#define NPIX    (NB * HW2)           // 100352
#define KDIM    (CIN * 9)            // 1152, reordered: k' = tap*128 + ci
#define KSTEP   32
#define NSTEPS  (KDIM / KSTEP)       // 36
#define TILE_M  128                  // c_out tile
#define TILE_N  128                  // pixel tile
#define LDSS    40                   // 32 + 8 pad halves (80B row stride, 16B aligned)
#define TILEH   (TILE_M * LDSS)      // halves per tile buffer (A and B same size)

#define AS1 __attribute__((address_space(1)))
#define AS3 __attribute__((address_space(3)))

#if defined(__gfx1250__) && __has_builtin(__builtin_amdgcn_global_load_async_to_lds_b128) && \
    __has_builtin(__builtin_amdgcn_s_wait_asynccnt)
#define HAVE_ASYNC_LDS 1
#else
#define HAVE_ASYNC_LDS 0
#endif

// ---- prep: weights OIHW f32 [co][ci*9+tap] -> f16 [co][tap*128+ci] ----
__global__ __launch_bounds__(256)
void prep_weights(const float* __restrict__ w, _Float16* __restrict__ wh) {
    const int idx = blockIdx.x * 256 + threadIdx.x;   // 256*1152 = 294912 total
    const int co  = idx / KDIM;
    const int r   = idx - co * KDIM;
    const int tap = r >> 7;          // r / 128
    const int ci  = r & 127;
    wh[idx] = (_Float16)w[co * KDIM + ci * 9 + tap];
}

__global__ __launch_bounds__(256, 2)
void conv3x3_wmma_v3(const float* __restrict__ x,
                     const _Float16* __restrict__ wh,
                     const float* __restrict__ bias,
                     float* __restrict__ out)
{
    __shared__ _Float16 sA[2 * TILEH];   // weights  [buf][co ][k32] f16
    __shared__ _Float16 sB[2 * TILEH];   // im2col   [buf][pix][k32] f16

    const int t       = threadIdx.x;
    const int lane    = t & 31;
    const int wave    = t >> 5;
    const int wm      = wave & 1;            // M half: 0 / 64
    const int wn      = wave >> 1;           // N quarter: 0/32/64/96
    const int lane16  = lane & 15;
    const int halfsel = lane >> 4;           // 0 or 1

    const int mBase = blockIdx.y * TILE_M;   // c_out base
    const int nBase = blockIdx.x * TILE_N;   // pixel base

    // staging assignment: 256 threads = 128 rows x 2 half-K-slices
    const int srow = t & 127;                // tile row (co for A, pixel for B)
    const int koff = (t >> 7) * 16;          // 0 or 16 within the 32-wide K step

    // per-thread pixel decomposition (once)
    const int p   = nBase + srow;
    const int pb  = p / HW2;
    const int phw = p - pb * HW2;
    const int ph  = phw / HW;
    const int pw  = phw - ph * HW;
    const float* __restrict__ xb = x + (size_t)pb * CIN * HW2;

    // A staging source row (already f16, tap-major K)
    const _Float16* __restrict__ wrow = wh + (size_t)(mBase + srow) * KDIM + koff;

    v8f acc[4][2];
    #pragma unroll
    for (int mi = 0; mi < 4; ++mi)
        #pragma unroll
        for (int nj = 0; nj < 2; ++nj)
            acc[mi][nj] = (v8f)0.0f;

    // ---- stage K-step `ks` (k' base = ks*32) into LDS buffer `buf` ----
    auto stage = [&](int ks, int buf) {
        const int tap = ks >> 2;                 // 4 ci-chunks per tap
        const int cs  = ks & 3;
        const int kh  = (tap * 11) >> 5;         // tap / 3 for tap in [0,8]
        const int kw  = tap - kh * 3;
        const int hh  = ph + kh - 1;
        const int ww  = pw + kw - 1;
        const bool valid = ((unsigned)hh < (unsigned)HW) &
                           ((unsigned)ww < (unsigned)HW);
        const int kbase = ks * KSTEP;
        const int bofs  = buf * TILEH;

        // A: pure f16 byte-copy via async global->LDS
        {
            const _Float16* src = wrow + kbase;
            _Float16*       dst = &sA[bofs + srow * LDSS + koff];
#if HAVE_ASYNC_LDS
            __builtin_amdgcn_global_load_async_to_lds_b128(
                (AS1 g4i*)(src),     (AS3 g4i*)(dst),     0, 0);
            __builtin_amdgcn_global_load_async_to_lds_b128(
                (AS1 g4i*)(src + 8), (AS3 g4i*)(dst + 8), 0, 0);
#else
            v8h a0 = *(const v8h*)(src);
            v8h a1 = *(const v8h*)(src + 8);
            *(v8h*)(dst)     = a0;
            *(v8h*)(dst + 8) = a1;
#endif
            if (ks + 1 < NSTEPS)
                __builtin_prefetch(wrow + kbase + KSTEP, 0, 1); // global_prefetch_b8
        }

        // B: 16 ci-strided f32 loads, one uniform pad branch, packed f16 stores
        {
            const int ci0 = cs * 32 + koff;
            float vals[16];
            if (valid) {
                const float* src = xb + (hh * HW + ww) + (size_t)ci0 * HW2;
                #pragma unroll
                for (int q = 0; q < 16; ++q)
                    vals[q] = src[(size_t)q * HW2];
            } else {
                #pragma unroll
                for (int q = 0; q < 16; ++q)
                    vals[q] = 0.0f;
            }
            v8h h0, h1;
            #pragma unroll
            for (int q = 0; q < 8; ++q) {
                h0[q] = (_Float16)vals[q];           // -> v_cvt_pk_f16_f32 pairs
                h1[q] = (_Float16)vals[8 + q];
            }
            _Float16* dst = &sB[bofs + srow * LDSS + koff];
            *(v8h*)(dst)     = h0;
            *(v8h*)(dst + 8) = h1;
        }
    };

    // ---- compute one K-step from LDS buffer `buf` ----
    auto compute = [&](int buf) {
        const int bofs = buf * TILEH;
        // batch all 12 ds_load_b128 first, then issue 8 back-to-back WMMAs
        v16h bfrag[2];
        #pragma unroll
        for (int nj = 0; nj < 2; ++nj) {
            const _Float16* src =
                &sB[bofs + (wn * 32 + nj * 16 + lane16) * LDSS + halfsel * 16];
            v8h lo = *(const v8h*)(src);
            v8h hi = *(const v8h*)(src + 8);
            bfrag[nj] = __builtin_shufflevector(lo, hi,
                0, 1, 2, 3, 4, 5, 6, 7, 8, 9, 10, 11, 12, 13, 14, 15);
        }
        v16h afrag[4];
        #pragma unroll
        for (int mi = 0; mi < 4; ++mi) {
            const _Float16* src =
                &sA[bofs + (wm * 64 + mi * 16 + lane16) * LDSS + halfsel * 16];
            v8h lo = *(const v8h*)(src);
            v8h hi = *(const v8h*)(src + 8);
            afrag[mi] = __builtin_shufflevector(lo, hi,
                0, 1, 2, 3, 4, 5, 6, 7, 8, 9, 10, 11, 12, 13, 14, 15);
        }
        #pragma unroll
        for (int mi = 0; mi < 4; ++mi)
            #pragma unroll
            for (int nj = 0; nj < 2; ++nj)
                acc[mi][nj] = __builtin_amdgcn_wmma_f32_16x16x32_f16(
                    false, afrag[mi], false, bfrag[nj],
                    (short)0, acc[mi][nj], false, false);
    };

    // ---- double-buffered main loop: one barrier per K-step ----
    stage(0, 0);
#if HAVE_ASYNC_LDS
    __builtin_amdgcn_s_wait_asynccnt(0);
#endif
    __syncthreads();

    for (int ks = 0; ks < NSTEPS; ++ks) {
        const int cur = ks & 1;
        if (ks + 1 < NSTEPS)
            stage(ks + 1, cur ^ 1);   // fill other buffer while computing this one
        compute(cur);
#if HAVE_ASYNC_LDS
        __builtin_amdgcn_s_wait_asynccnt(0);   // async A-copies for next step done
#endif
        __syncthreads();                        // publish next buffer / retire cur
    }

    // ---- epilogue: bias + coalesced NCHW store ----
    #pragma unroll
    for (int nj = 0; nj < 2; ++nj) {
        const int n   = nBase + wn * 32 + nj * 16 + lane16;
        const int nb  = n / HW2;
        const int nhw = n - nb * HW2;
        float* __restrict__ obase = out + (size_t)nb * COUT * HW2 + nhw;
        #pragma unroll
        for (int mi = 0; mi < 4; ++mi) {
            const v8f a = acc[mi][nj];
            const int co0 = mBase + wm * 64 + mi * 16 + halfsel * 8;
            #pragma unroll
            for (int r = 0; r < 8; ++r) {
                const int co = co0 + r;
                obase[(size_t)co * HW2] = a[r] + bias[co];
            }
        }
    }
}

extern "C" void kernel_launch(void* const* d_in, const int* in_sizes, int n_in,
                              void* d_out, int out_size, void* d_ws, size_t ws_size,
                              hipStream_t stream) {
    (void)in_sizes; (void)n_in; (void)out_size; (void)ws_size;
    const float* x    = (const float*)d_in[0];   // [32,128,56,56]
    const float* wgt  = (const float*)d_in[1];   // [256,128,3,3]
    const float* bias = (const float*)d_in[2];   // [256]
    float*       out  = (float*)d_out;           // [32,256,56,56]
    _Float16*    wh   = (_Float16*)d_ws;         // [256][9][128] f16 = 576 KB

    prep_weights<<<dim3(COUT * KDIM / 256), dim3(256), 0, stream>>>(wgt, wh);

    dim3 grid(NPIX / TILE_N, COUT / TILE_M, 1);  // 784 x 2
    dim3 block(256, 1, 1);
    conv3x3_wmma_v3<<<grid, block, 0, stream>>>(x, wh, bias, out);
}
